// MSSA_62637803045422
// MI455X (gfx1250) — matrix operand order repaired and checked by reading, not compiled, not run on hardware
//
#include <hip/hip_runtime.h>
#include <hip/hip_bf16.h>
#include <math.h>

// ---------------------------------------------------------------------------
// MSSA forward for MI455X (gfx1250).
// All contractions run on v_wmma_f32_16x16x32_bf16 with double-buffered LDS
// tiles, b128 LDS traffic, fused epilogues, and guard-free loaders (inputs
// padded so K % 32 == 0 and concat boundaries are chunk-aligned).
// Geometry fixed by the reference: B=16, C=128, H=W=64.
// ---------------------------------------------------------------------------

#define BATCH 16
#define CCH   128
#define HIMG  64
#define WIMG  64
#define HWIMG 4096          // 64*64
#define AUXP  288           // aux channels padded 277 -> 288 (9 k-steps)

typedef __attribute__((ext_vector_type(16))) __bf16 bf16x16;
typedef __attribute__((ext_vector_type(8)))  float  floatx8;

union FragAB { bf16x16 v; uint4 q[2]; };
union AccU   { floatx8 v; float f[8]; };

static __device__ __forceinline__ __bf16 to_bf16(float x) {
    return static_cast<__bf16>(x);
}

// ---------------------------------------------------------------------------
// Wave-level WMMA micro-step.
// LDS: As[128][32] bf16 (rows = M), Bs[64][32] bf16 (rows = N, K-major).
// Each wave owns a 16(M) x 64(N) tile => 4 accumulators of 16x16 f32.
// Fragment layouts per CDNA5 ISA 7.12.2:
//   A (16x32 bf16): m=lane&15; K chunks [hi*8, +8) and [hi*8+16, +8)
//   B (32x16 bf16): n=lane&15; K chunks [hi*8, +8) and [hi*8+16, +8)
//   D (16x16 f32):  vgpr r -> m = r + hi*8; n = lane&15
// ---------------------------------------------------------------------------
static __device__ __forceinline__ void wmma_tile(const __bf16* __restrict__ As,
                                                 const __bf16* __restrict__ Bs,
                                                 AccU* acc, int m0, int lane) {
    const int row = lane & 15;
    const int hi  = lane >> 4;
    FragAB fa;
    const __bf16* ap = As + (size_t)(m0 + row) * 32;
    fa.q[0] = *(const uint4*)(ap + hi * 8);
    fa.q[1] = *(const uint4*)(ap + hi * 8 + 16);
#pragma unroll
    for (int j = 0; j < 4; ++j) {
        FragAB fb;
        const __bf16* bp = Bs + (size_t)(j * 16 + row) * 32 + hi * 8;
        fb.q[0] = *(const uint4*)(bp);
        fb.q[1] = *(const uint4*)(bp + 16);
        acc[j].v = __builtin_amdgcn_wmma_f32_16x16x32_bf16(
            false, fa.v, false, fb.v, (short)0, acc[j].v, false, false);
    }
}

// ---------------------------------------------------------------------------
// One-time packing: aux = concat(edge, seg, pred_edge, pred_seg) -> [B,288,HW]
// f32, zero-padded channels 277..287. One (channel, batch) per block.
// ---------------------------------------------------------------------------
__global__ __launch_bounds__(256) void k_pack_aux(
    const float* __restrict__ e, const float* __restrict__ s,
    const float* __restrict__ pe, const float* __restrict__ ps,
    float* __restrict__ out)
{
    const int k = blockIdx.x, b = blockIdx.y, tid = threadIdx.x;
    float4* o4 = (float4*)(out + ((size_t)b * AUXP + k) * HWIMG);
    const float* src = nullptr; int kl = 0, kc = 0;
    if (k < 128)      { src = e;  kl = k;       kc = 128; }
    else if (k < 256) { src = s;  kl = k - 128; kc = 128; }
    else if (k < 257) { src = pe; kl = 0;       kc = 1;   }
    else if (k < 277) { src = ps; kl = k - 257; kc = 20;  }
    if (src) {
        const float4* s4 = (const float4*)(src + ((size_t)b * kc + kl) * HWIMG);
        for (int i = tid; i < HWIMG / 4; i += 256) o4[i] = s4[i];
    } else {
        const float4 z = {0.f, 0.f, 0.f, 0.f};
        for (int i = tid; i < HWIMG / 4; i += 256) o4[i] = z;
    }
}

// Pad embed weights [128,277] -> [128,288] (zeros beyond 277).
__global__ __launch_bounds__(256) void k_pad_w(const float* __restrict__ w,
                                               float* __restrict__ out)
{
    const int m = blockIdx.x;
    for (int k = threadIdx.x; k < AUXP; k += 256)
        out[(size_t)m * AUXP + k] = (k < 277) ? w[(size_t)m * 277 + k] : 0.f;
}

// ---------------------------------------------------------------------------
// 1x1 conv as GEMM: out[b,o,p] = sum_k W[o,k] * in(k)[b,p] + bias[o]
// Requirements (guaranteed by caller): K % 32 == 0, part sizes multiples of
// 8 (so a chunk of 8 never crosses a part boundary). O fixed to 128.
// Grid: (HW/64, 1, B), block 256 (8 waves).
// ---------------------------------------------------------------------------
__global__ __launch_bounds__(256) void k_conv1x1(
    const float* __restrict__ Wt, const float* __restrict__ Bi,
    const float* __restrict__ in0, const float* __restrict__ in1,
    int k0, int K,
    float* __restrict__ out)
{
    __shared__ __align__(16) __bf16 As[2][128 * 32];
    __shared__ __align__(16) __bf16 Bs[2][64 * 32];
    const int tid  = threadIdx.x;
    const int b    = blockIdx.z;
    const int p0   = blockIdx.x * 64;
    const int wave = tid >> 5, lane = tid & 31;
    const int nT   = K >> 5;

    AccU acc[4];
#pragma unroll
    for (int j = 0; j < 4; ++j)
#pragma unroll
        for (int r = 0; r < 8; ++r) acc[j].f[r] = 0.f;

    // A tile: 2 chunks of 8 per thread; float4 pair + one ds_store_b128.
    auto fillA = [&](int s, int kt) {
#pragma unroll
        for (int e = 0; e < 2; ++e) {
            int chunk = e * 256 + tid;
            int m = chunk >> 2, ko = (chunk & 3) * 8;
            const float4 f0 = *(const float4*)&Wt[(size_t)m * K + kt + ko];
            const float4 f1 = *(const float4*)&Wt[(size_t)m * K + kt + ko + 4];
            alignas(16) __bf16 t[8];
            t[0] = to_bf16(f0.x); t[1] = to_bf16(f0.y);
            t[2] = to_bf16(f0.z); t[3] = to_bf16(f0.w);
            t[4] = to_bf16(f1.x); t[5] = to_bf16(f1.y);
            t[6] = to_bf16(f1.z); t[7] = to_bf16(f1.w);
            *(uint4*)&As[s][m * 32 + ko] = *(const uint4*)t;
        }
    };
    // B tile: one chunk of 8 channels per thread; part-select once per chunk;
    // 8 unconditional coalesced loads + one ds_store_b128.
    auto fillB = [&](int s, int kt) {
        const int prow = tid & 63, ko = (tid >> 6) * 8;
        int kl = kt + ko;
        const float* src = in0;
        if (kl >= k0) { src = in1; kl -= k0; }
        const float* base = src + ((size_t)b * ((src == in0) ? k0 : (K - k0)) + kl) * HWIMG
                          + p0 + prow;
        alignas(16) __bf16 t[8];
#pragma unroll
        for (int j2 = 0; j2 < 8; ++j2)
            t[j2] = to_bf16(base[(size_t)j2 * HWIMG]);
        *(uint4*)&Bs[s][prow * 32 + ko] = *(const uint4*)t;
    };

    fillA(0, 0); fillB(0, 0);
    __syncthreads();
    for (int t = 0; t < nT; ++t) {
        const int cur = t & 1;
        if (t + 1 < nT) { fillA(cur ^ 1, (t + 1) * 32); fillB(cur ^ 1, (t + 1) * 32); }
        wmma_tile(&As[cur][0], &Bs[cur][0], acc, wave * 16, lane);
        __syncthreads();
    }

    const int hi = lane >> 4, col = lane & 15;
#pragma unroll
    for (int j = 0; j < 4; ++j) {
        int p = p0 + j * 16 + col;
#pragma unroll
        for (int r = 0; r < 8; ++r) {
            int o = wave * 16 + r + hi * 8;
            out[((size_t)b * CCH + o) * HWIMG + p] = acc[j].f[r] + Bi[o];
        }
    }
}

// ---------------------------------------------------------------------------
// LayerNorm over [C,H,W] per sample.
// ---------------------------------------------------------------------------
__global__ __launch_bounds__(256) void k_ln_stats(const float* __restrict__ xin,
                                                  size_t n, float* __restrict__ stats)
{
    const int tid = threadIdx.x;
    const float4* p4 = (const float4*)(xin + (size_t)blockIdx.x * n);
    const size_t n4 = n >> 2;
    float s = 0.f, ss = 0.f;
    for (size_t i = tid; i < n4; i += 256) {
        float4 v = p4[i];
        s  += v.x + v.y + v.z + v.w;
        ss += v.x * v.x + v.y * v.y + v.z * v.z + v.w * v.w;
    }
    __shared__ float sh1[256], sh2[256];
    sh1[tid] = s; sh2[tid] = ss;
    __syncthreads();
    for (int st = 128; st > 0; st >>= 1) {
        if (tid < st) { sh1[tid] += sh1[tid + st]; sh2[tid] += sh2[tid + st]; }
        __syncthreads();
    }
    if (tid == 0) {
        float mu  = sh1[0] / (float)n;
        float var = sh2[0] / (float)n - mu * mu;
        stats[blockIdx.x * 2 + 0] = mu;
        stats[blockIdx.x * 2 + 1] = rsqrtf(var + 1e-5f);
    }
}

// out = (x - mu)*rstd * gamma + beta   (vectorized, in-place safe)
__global__ __launch_bounds__(256) void k_ln_apply(
    const float* __restrict__ xin, const float* __restrict__ g,
    const float* __restrict__ be, const float* __restrict__ stats,
    float* __restrict__ out, size_t n)
{
    const size_t b = blockIdx.y;
    const float mu = stats[b * 2 + 0];
    const float rs = stats[b * 2 + 1];
    const size_t n4 = n >> 2;
    const float4* x4 = (const float4*)(xin + b * n);
    const float4* g4 = (const float4*)(g + b * n);
    const float4* b4 = (const float4*)(be + b * n);
    float4*       o4 = (float4*)(out + b * n);
    const size_t stride = (size_t)gridDim.x * 256;
    for (size_t i = (size_t)blockIdx.x * 256 + threadIdx.x; i < n4; i += stride) {
        float4 xv = x4[i], gv = g4[i], bv = b4[i], ov;
        ov.x = (xv.x - mu) * rs * gv.x + bv.x;
        ov.y = (xv.y - mu) * rs * gv.y + bv.y;
        ov.z = (xv.z - mu) * rs * gv.z + bv.z;
        ov.w = (xv.w - mu) * rs * gv.w + bv.w;
        o4[i] = ov;
    }
}

// ---------------------------------------------------------------------------
// Window attention, templated on the patch geometry so all index math is
// shifts/masks. Token t = wy*OW+wx; feature d = c*(PH*PW) + py*PW + px.
// ---------------------------------------------------------------------------
template<int PH, int PW, int N, int D, int OW>
__global__ __launch_bounds__(256) void k_attn_scores(
    const float* __restrict__ Q, const float* __restrict__ Kt,
    float* __restrict__ S, int cbase)
{
    __shared__ __align__(16) __bf16 As[2][128 * 32];
    __shared__ __align__(16) __bf16 Bs[2][64 * 32];
    const int tid  = threadIdx.x;
    const int b    = blockIdx.z;
    const int i0   = blockIdx.y * 128;
    const int j0   = blockIdx.x * 64;
    const int wave = tid >> 5, lane = tid & 31;
    constexpr int PP = PH * PW;
    constexpr int nT = D / 32;
    const float scale = 1.0f / sqrtf((float)D);

    AccU acc[4];
#pragma unroll
    for (int j = 0; j < 4; ++j)
#pragma unroll
        for (int r = 0; r < 8; ++r) acc[j].f[r] = 0.f;

    auto gatherTok = [&](const float* src, int tok, int dbase, __bf16* t) {
        const int wy = tok / OW, wx = tok % OW;       // OW power of two
        const float* base = src + ((size_t)b * CCH + cbase) * HWIMG
                          + (size_t)(wy * PH) * WIMG + wx * PW;
#pragma unroll
        for (int j2 = 0; j2 < 8; ++j2) {
            int d = dbase + j2;
            int c = d / PP, rr = d % PP, py = rr / PW, px = rr % PW;
            t[j2] = to_bf16(base[(size_t)c * HWIMG + py * WIMG + px]);
        }
    };
    auto fillQ = [&](int s, int dt) {
#pragma unroll
        for (int e = 0; e < 2; ++e) {
            int chunk = e * 256 + tid;
            int il = chunk & 127, ko = (chunk >> 7) * 8;
            alignas(16) __bf16 t[8];
            gatherTok(Q, i0 + il, dt + ko, t);
            *(uint4*)&As[s][il * 32 + ko] = *(const uint4*)t;
        }
    };
    auto fillK = [&](int s, int dt) {
        int jl = tid & 63, ko = (tid >> 6) * 8;
        alignas(16) __bf16 t[8];
        gatherTok(Kt, j0 + jl, dt + ko, t);
        *(uint4*)&Bs[s][jl * 32 + ko] = *(const uint4*)t;
    };

    fillQ(0, 0); fillK(0, 0);
    __syncthreads();
    for (int t = 0; t < nT; ++t) {
        const int cur = t & 1;
        if (t + 1 < nT) { fillQ(cur ^ 1, (t + 1) * 32); fillK(cur ^ 1, (t + 1) * 32); }
        wmma_tile(&As[cur][0], &Bs[cur][0], acc, wave * 16, lane);
        __syncthreads();
    }

    const int hi = lane >> 4, col = lane & 15;
#pragma unroll
    for (int j = 0; j < 4; ++j) {
        int jg = j0 + j * 16 + col;
#pragma unroll
        for (int r = 0; r < 8; ++r) {
            int ig = i0 + wave * 16 + r + hi * 8;
            S[((size_t)b * N + ig) * N + jg] = acc[j].f[r] * scale;
        }
    }
}

// Row softmax, one block per (b,row), float4 passes.
__global__ __launch_bounds__(256) void k_softmax(float* __restrict__ S, int N)
{
    const int tid = threadIdx.x;
    float4* row4 = (float4*)(S + ((size_t)blockIdx.y * N + blockIdx.x) * N);
    const int N4 = N >> 2;
    __shared__ float sh[256];
    float mx = -3.4e38f;
    for (int i = tid; i < N4; i += 256) {
        float4 v = row4[i];
        mx = fmaxf(mx, fmaxf(fmaxf(v.x, v.y), fmaxf(v.z, v.w)));
    }
    sh[tid] = mx; __syncthreads();
    for (int s = 128; s > 0; s >>= 1) { if (tid < s) sh[tid] = fmaxf(sh[tid], sh[tid + s]); __syncthreads(); }
    mx = sh[0];
    __syncthreads();
    float sum = 0.f;
    for (int i = tid; i < N4; i += 256) {
        float4 v = row4[i];
        v.x = __expf(v.x - mx); v.y = __expf(v.y - mx);
        v.z = __expf(v.z - mx); v.w = __expf(v.w - mx);
        row4[i] = v;
        sum += v.x + v.y + v.z + v.w;
    }
    sh[tid] = sum; __syncthreads();
    for (int s = 128; s > 0; s >>= 1) { if (tid < s) sh[tid] += sh[tid + s]; __syncthreads(); }
    float inv = 1.f / sh[0];
    for (int i = tid; i < N4; i += 256) {
        float4 v = row4[i];
        v.x *= inv; v.y *= inv; v.z *= inv; v.w *= inv;
        row4[i] = v;
    }
}

// P x V with residual fused: mid[...] = x[...] + sum_j P[i,j] V[j,d]
template<int PH, int PW, int N, int D, int OW>
__global__ __launch_bounds__(256) void k_attn_av(
    const float* __restrict__ P, const float* __restrict__ V,
    const float* __restrict__ xres, float* __restrict__ mid, int cbase)
{
    __shared__ __align__(16) __bf16 As[2][128 * 32];
    __shared__ __align__(16) __bf16 Bs[2][64 * 32];
    const int tid  = threadIdx.x;
    const int b    = blockIdx.z;
    const int i0   = blockIdx.y * 128;
    const int d0   = blockIdx.x * 64;
    const int wave = tid >> 5, lane = tid & 31;
    constexpr int PP = PH * PW;
    constexpr int nT = N / 32;

    AccU acc[4];
#pragma unroll
    for (int j = 0; j < 4; ++j)
#pragma unroll
        for (int r = 0; r < 8; ++r) acc[j].f[r] = 0.f;

    auto fillP = [&](int s, int jt) {   // contiguous rows -> float4 loads
#pragma unroll
        for (int e = 0; e < 2; ++e) {
            int chunk = e * 256 + tid;
            int il = chunk >> 2, jo = (chunk & 3) * 8;
            const float* rp = P + ((size_t)b * N + i0 + il) * N + jt + jo;
            const float4 f0 = *(const float4*)rp;
            const float4 f1 = *(const float4*)(rp + 4);
            alignas(16) __bf16 t[8];
            t[0] = to_bf16(f0.x); t[1] = to_bf16(f0.y);
            t[2] = to_bf16(f0.z); t[3] = to_bf16(f0.w);
            t[4] = to_bf16(f1.x); t[5] = to_bf16(f1.y);
            t[6] = to_bf16(f1.z); t[7] = to_bf16(f1.w);
            *(uint4*)&As[s][il * 32 + jo] = *(const uint4*)t;
        }
    };
    auto fillV = [&](int s, int jt) {   // fixed d per thread, 8 token gather
        int dl = tid & 63, jo = (tid >> 6) * 8;
        int d = d0 + dl;
        int c = d / PP, rr = d % PP, py = rr / PW, px = rr % PW;
        const float* base = V + ((size_t)b * CCH + cbase + c) * HWIMG
                          + (size_t)py * WIMG + px;
        alignas(16) __bf16 t[8];
#pragma unroll
        for (int j2 = 0; j2 < 8; ++j2) {
            int tok = jt + jo + j2;
            int wy = tok / OW, wx = tok % OW;
            t[j2] = to_bf16(base[(size_t)(wy * PH) * WIMG + wx * PW]);
        }
        *(uint4*)&Bs[s][dl * 32 + jo] = *(const uint4*)t;
    };

    fillP(0, 0); fillV(0, 0);
    __syncthreads();
    for (int t = 0; t < nT; ++t) {
        const int cur = t & 1;
        if (t + 1 < nT) { fillP(cur ^ 1, (t + 1) * 32); fillV(cur ^ 1, (t + 1) * 32); }
        wmma_tile(&As[cur][0], &Bs[cur][0], acc, wave * 16, lane);
        __syncthreads();
    }

    const int hi = lane >> 4, col = lane & 15;
#pragma unroll
    for (int j = 0; j < 4; ++j) {
        int d = d0 + j * 16 + col;
        int c = d / PP, rr = d % PP, py = rr / PW, px = rr % PW;
#pragma unroll
        for (int r = 0; r < 8; ++r) {
            int ig = i0 + wave * 16 + r + hi * 8;
            int wy = ig / OW, wx = ig % OW;
            size_t gi = ((size_t)b * CCH + cbase + c) * HWIMG + (wy * PH + py) * WIMG + wx * PW + px;
            mid[gi] = xres[gi] + acc[j].f[r];
        }
    }
}

// ---------------------------------------------------------------------------
// 3x3 conv as implicit GEMM (K = 128*9 = 1152), im2col gather in the B-loader.
// mode: 1 = tanh, 2 = exact GELU, 3 = out = mid*gamma + (acc+bias)
// Grid: (HW/64, 1, B).
// ---------------------------------------------------------------------------
__global__ __launch_bounds__(256) void k_conv3x3(
    const float* __restrict__ in, const float* __restrict__ Wt,
    const float* __restrict__ Bi, const float* __restrict__ midp,
    const float* __restrict__ gammap, float* __restrict__ out,
    int dil, int mode)
{
    __shared__ __align__(16) __bf16 As[2][128 * 32];
    __shared__ __align__(16) __bf16 Bs[2][64 * 32];
    const int tid  = threadIdx.x;
    const int b    = blockIdx.z;
    const int p0   = blockIdx.x * 64;
    const int wave = tid >> 5, lane = tid & 31;
    const int K = CCH * 9;   // 1152
    const int nT = K >> 5;

    AccU acc[4];
#pragma unroll
    for (int j = 0; j < 4; ++j)
#pragma unroll
        for (int r = 0; r < 8; ++r) acc[j].f[r] = 0.f;

    auto fillA = [&](int s, int kt) {
#pragma unroll
        for (int e = 0; e < 2; ++e) {
            int chunk = e * 256 + tid;
            int m = chunk >> 2, ko = (chunk & 3) * 8;
            const float4 f0 = *(const float4*)&Wt[(size_t)m * K + kt + ko];
            const float4 f1 = *(const float4*)&Wt[(size_t)m * K + kt + ko + 4];
            alignas(16) __bf16 t[8];
            t[0] = to_bf16(f0.x); t[1] = to_bf16(f0.y);
            t[2] = to_bf16(f0.z); t[3] = to_bf16(f0.w);
            t[4] = to_bf16(f1.x); t[5] = to_bf16(f1.y);
            t[6] = to_bf16(f1.z); t[7] = to_bf16(f1.w);
            *(uint4*)&As[s][m * 32 + ko] = *(const uint4*)t;
        }
    };
    auto fillB = [&](int s, int kt) {
        const int prow = tid & 63, ko = (tid >> 6) * 8;
        const int p = p0 + prow;
        const int h = p >> 6, wc = p & 63;
        alignas(16) __bf16 t[8];
#pragma unroll
        for (int j2 = 0; j2 < 8; ++j2) {
            int kg = kt + ko + j2;
            int c = kg / 9, t9 = kg % 9;            // constant divisor -> mul
            int ky = t9 / 3 - 1, kx = t9 % 3 - 1;
            int hh = h + ky * dil, ww = wc + kx * dil;
            float v = 0.f;
            if (hh >= 0 && hh < HIMG && ww >= 0 && ww < WIMG)
                v = in[((size_t)b * CCH + c) * HWIMG + hh * WIMG + ww];
            t[j2] = to_bf16(v);
        }
        *(uint4*)&Bs[s][prow * 32 + ko] = *(const uint4*)t;
    };

    fillA(0, 0); fillB(0, 0);
    __syncthreads();
    for (int t = 0; t < nT; ++t) {
        const int cur = t & 1;
        if (t + 1 < nT) { fillA(cur ^ 1, (t + 1) * 32); fillB(cur ^ 1, (t + 1) * 32); }
        wmma_tile(&As[cur][0], &Bs[cur][0], acc, wave * 16, lane);
        __syncthreads();
    }

    const int hi = lane >> 4, col = lane & 15;
#pragma unroll
    for (int j = 0; j < 4; ++j) {
        int p = p0 + j * 16 + col;
#pragma unroll
        for (int r = 0; r < 8; ++r) {
            int o = wave * 16 + r + hi * 8;
            size_t gi = ((size_t)b * CCH + o) * HWIMG + p;
            float v = acc[j].f[r] + Bi[o];
            if (mode == 1)      v = tanhf(v);
            else if (mode == 2) v = 0.5f * v * (1.f + erff(v * 0.70710678118654752f));
            else if (mode == 3) v = midp[gi] * gammap[gi] + v;
            out[gi] = v;
        }
    }
}

// ---------------------------------------------------------------------------
// Orchestration
// ---------------------------------------------------------------------------
extern "C" void kernel_launch(void* const* d_in, const int* in_sizes, int n_in,
                              void* d_out, int out_size, void* d_ws, size_t ws_size,
                              hipStream_t stream) {
    (void)in_sizes; (void)n_in; (void)out_size; (void)ws_size;
    const float* x         = (const float*)d_in[0];
    const float* edge      = (const float*)d_in[1];
    const float* seg       = (const float*)d_in[2];
    const float* pred_edge = (const float*)d_in[3];
    const float* pred_seg  = (const float*)d_in[4];
    const float* q_w = (const float*)d_in[5];  const float* q_b = (const float*)d_in[6];
    const float* k_w = (const float*)d_in[7];  const float* k_b = (const float*)d_in[8];
    const float* v_w = (const float*)d_in[9];  const float* v_b = (const float*)d_in[10];
    const float* d1_ew = (const float*)d_in[11]; const float* d1_eb = (const float*)d_in[12];
    const float* d1_sw = (const float*)d_in[13]; const float* d1_sb = (const float*)d_in[14];
    const float* d1_bw = (const float*)d_in[15]; const float* d1_bb = (const float*)d_in[16];
    const float* d2_ew = (const float*)d_in[17]; const float* d2_eb = (const float*)d_in[18];
    const float* d2_sw = (const float*)d_in[19]; const float* d2_sb = (const float*)d_in[20];
    const float* d2_bw = (const float*)d_in[21]; const float* d2_bb = (const float*)d_in[22];
    const float* sc_w  = (const float*)d_in[23]; const float* sc_b  = (const float*)d_in[24];
    const float* bi1_w = (const float*)d_in[25]; const float* bi1_b = (const float*)d_in[26];
    const float* bi2_w = (const float*)d_in[27]; const float* bi2_b = (const float*)d_in[28];
    float* outp = (float*)d_out;
    float* ws   = (float*)d_ws;

    const size_t T = (size_t)BATCH * CCH * HWIMG;        // 8,388,608 floats
    float* bufQ   = ws;
    float* bufK   = ws + 1 * T;
    float* bufV   = ws + 2 * T;
    float* bufA   = ws + 3 * T;   // shared embed a (d1 then d2)
    float* bufG   = ws + 4 * T;   // shared gamma (d1 then d2)
    float* bufBe  = ws + 5 * T;   // per-tensor beta (reused sequentially)
    float* bufMid = ws + 6 * T;
    float* scores = ws + 7 * T;                          // up to 16M floats
    float* stats  = scores + (size_t)16 * 1024 * 1024;   // 32 floats
    float* bufAux = stats + 64;                          // [B,288,HW] padded aux
    float* bufW1  = bufAux + (size_t)BATCH * AUXP * HWIMG;  // [128,288]
    float* bufW2  = bufW1 + (size_t)CCH * AUXP;

    const dim3 blk(256);
    const dim3 g64(HWIMG / 64, 1, BATCH);
    const size_t nPer = (size_t)CCH * HWIMG;             // LN extent per sample

    // one-time padding/packing (guard-free GEMMs afterwards)
    k_pack_aux<<<dim3(AUXP, BATCH), blk, 0, stream>>>(edge, seg, pred_edge, pred_seg, bufAux);
    k_pad_w<<<dim3(CCH), blk, 0, stream>>>(d1_ew, bufW1);
    k_pad_w<<<dim3(CCH), blk, 0, stream>>>(d2_ew, bufW2);

    // q/k/v projections
    k_conv1x1<<<g64, blk, 0, stream>>>(q_w, q_b, x, nullptr, 128, 128, bufQ);
    k_conv1x1<<<g64, blk, 0, stream>>>(k_w, k_b, x, nullptr, 128, 128, bufK);
    k_conv1x1<<<g64, blk, 0, stream>>>(v_w, v_b, x, nullptr, 128, 128, bufV);
    // shared aux embed + gamma (same for q,k,v)
    k_conv1x1<<<g64, blk, 0, stream>>>(bufW1, d1_eb, bufAux, nullptr, AUXP, AUXP, bufA);
    k_conv1x1<<<g64, blk, 0, stream>>>(d1_sw, d1_sb, bufA, nullptr, 128, 128, bufG);

    // ADN per tensor: beta = bias_w . [proj, a]; then LN-apply in place
    k_conv1x1<<<g64, blk, 0, stream>>>(d1_bw, d1_bb, bufQ, bufA, 128, 256, bufBe);
    k_ln_stats<<<BATCH, blk, 0, stream>>>(bufQ, nPer, stats);
    k_ln_apply<<<dim3(512, BATCH), blk, 0, stream>>>(bufQ, bufG, bufBe, stats, bufQ, nPer);

    k_conv1x1<<<g64, blk, 0, stream>>>(d1_bw, d1_bb, bufK, bufA, 128, 256, bufBe);
    k_ln_stats<<<BATCH, blk, 0, stream>>>(bufK, nPer, stats);
    k_ln_apply<<<dim3(512, BATCH), blk, 0, stream>>>(bufK, bufG, bufBe, stats, bufK, nPer);

    k_conv1x1<<<g64, blk, 0, stream>>>(d1_bw, d1_bb, bufV, bufA, 128, 256, bufBe);
    k_ln_stats<<<BATCH, blk, 0, stream>>>(bufV, nPer, stats);
    k_ln_apply<<<dim3(512, BATCH), blk, 0, stream>>>(bufV, bufG, bufBe, stats, bufV, nPer);

    // attention, patch (2,2): channels 0..63, N=1024 tokens, d=256
    k_attn_scores<2, 2, 1024, 256, 32><<<dim3(16, 8, BATCH), blk, 0, stream>>>(bufQ, bufK, scores, 0);
    k_softmax<<<dim3(1024, BATCH), blk, 0, stream>>>(scores, 1024);
    k_attn_av<2, 2, 1024, 256, 32><<<dim3(4, 8, BATCH), blk, 0, stream>>>(scores, bufV, x, bufMid, 0);
    // attention, patch (4,4): channels 64..127, N=256 tokens, d=1024
    k_attn_scores<4, 4, 256, 1024, 16><<<dim3(4, 2, BATCH), blk, 0, stream>>>(bufQ, bufK, scores, 64);
    k_softmax<<<dim3(256, BATCH), blk, 0, stream>>>(scores, 256);
    k_attn_av<4, 4, 256, 1024, 16><<<dim3(16, 2, BATCH), blk, 0, stream>>>(scores, bufV, x, bufMid, 64);

    // second ADN on mid -> o (stored in bufQ, q no longer needed)
    k_conv1x1<<<g64, blk, 0, stream>>>(bufW2, d2_eb, bufAux, nullptr, AUXP, AUXP, bufA);
    k_conv1x1<<<g64, blk, 0, stream>>>(d2_sw, d2_sb, bufA, nullptr, 128, 128, bufG);
    k_conv1x1<<<g64, blk, 0, stream>>>(d2_bw, d2_bb, bufMid, bufA, 128, 256, bufBe);
    k_ln_stats<<<BATCH, blk, 0, stream>>>(bufMid, nPer, stats);
    k_ln_apply<<<dim3(512, BATCH), blk, 0, stream>>>(bufMid, bufG, bufBe, stats, bufQ, nPer);

    // final 3x3 convs: gamma = tanh(sc*o) -> bufK; t1 = gelu(bi1*o, dil=2) -> bufV;
    // out = mid*gamma + (bi2*t1 + b) fused into the last conv's epilogue.
    k_conv3x3<<<g64, blk, 0, stream>>>(bufQ, sc_w, sc_b, nullptr, nullptr, bufK, 1, 1);
    k_conv3x3<<<g64, blk, 0, stream>>>(bufQ, bi1_w, bi1_b, nullptr, nullptr, bufV, 2, 2);
    k_conv3x3<<<g64, blk, 0, stream>>>(bufV, bi2_w, bi2_b, bufMid, bufK, outp, 1, 3);
}
